// imageCaptionModel_40278203301943
// MI455X (gfx1250) — compile-verified, matrix operand order checked by reading, not run on hardware
//
#include <hip/hip_runtime.h>

// ---------------------------------------------------------------------------
// Types for CDNA5 WMMA (wave32): 16x16x32 bf16 -> f32
// ---------------------------------------------------------------------------
typedef __attribute__((ext_vector_type(16))) __bf16 v16bf;
typedef __attribute__((ext_vector_type(8)))  __bf16 v8bf;
typedef __attribute__((ext_vector_type(8)))  float  v8f;

// Model dims
#define BB   256
#define TT   20
#define VV   10000
#define EE   512
#define HH   1024
#define NFF  2048
#define LL   10
#define MM   512
#define K0   2048   // E+M+H
#define K1P  2080   // H+10+H = 2058 padded to mult of 32

__device__ __forceinline__ float sigm(float x) { return 1.f / (1.f + __expf(-x)); }

// Load a 16x32 bf16 fragment (A-layout; B loaded from [N,K] row-major uses
// the identical striping).  lane 0-15: row = lane,   K = [0..8) u [16..24)
// lane 16-31: row = lane-16, K = [8..16) u [24..32).  (ISA 7.12.2)
__device__ __forceinline__ v16bf load_frag(const __bf16* __restrict__ base,
                                           int ld, int row, int k0, int lane) {
  const int m    = lane & 15;
  const int koff = (lane >> 4) << 3;
  const __bf16* p = base + (size_t)(row + m) * (size_t)ld + (size_t)(k0 + koff);
  v8bf lo = *reinterpret_cast<const v8bf*>(p);
  v8bf hi = *reinterpret_cast<const v8bf*>(p + 16);
  return __builtin_shufflevector(lo, hi, 0,1,2,3,4,5,6,7,8,9,10,11,12,13,14,15);
}

#define WMMA_BF16(A, Bm, C) \
  __builtin_amdgcn_wmma_f32_16x16x32_bf16(false, (A), false, (Bm), (short)0, (C), false, false)

// ---------------------------------------------------------------------------
// Prep kernels: fp32 -> bf16 conversions / transposes (one-time per launch)
// ---------------------------------------------------------------------------
__global__ void transpose_to_bf16_kernel(const float* __restrict__ W,
                                         __bf16* __restrict__ Wt,
                                         int K, int N, int Kpad) {
  size_t idx = (size_t)blockIdx.x * blockDim.x + threadIdx.x;
  size_t total = (size_t)N * Kpad;
  if (idx >= total) return;
  int n = (int)(idx / Kpad), k = (int)(idx % Kpad);
  Wt[idx] = (k < K) ? (__bf16)W[(size_t)k * N + n] : (__bf16)0.f;
}

__global__ void f32_to_bf16_kernel(const float* __restrict__ s,
                                   __bf16* __restrict__ d, size_t n) {
  size_t idx = (size_t)blockIdx.x * blockDim.x + threadIdx.x;
  if (idx < n) d[idx] = (__bf16)s[idx];
}

// cnn_features [B, NF, L] -> A_conv [(b*L+l), NF] bf16
__global__ void conv_a_prep_kernel(const float* __restrict__ cnn,
                                   __bf16* __restrict__ A) {
  size_t idx = (size_t)blockIdx.x * blockDim.x + threadIdx.x;
  if (idx >= (size_t)BB * LL * NFF) return;
  int row = (int)(idx / NFF), c = (int)(idx % NFF);
  int b = row / LL, l = row % LL;
  A[idx] = (__bf16)cnn[((size_t)b * NFF + c) * LL + l];
}

// ---------------------------------------------------------------------------
// Generic WMMA GEMM, register-blocked over M:
//   out[M,N] = A[M,K](bf16) * Bt[N,K](bf16)^T + bias
// Each wave: MBLK m-tiles x 1 n-tile; one B fragment feeds MBLK WMMAs
// (cuts B-matrix L2 traffic by MBLK; B is the dominant operand since N >> M).
// ---------------------------------------------------------------------------
template <int MBLK>
__global__ __launch_bounds__(256)
void gemm_bias_kernel(const __bf16* __restrict__ A, const __bf16* __restrict__ Bt,
                      const float* __restrict__ bias, float* __restrict__ out,
                      int ntiles_n, int total_tiles, int K, long out_ld) {
  const int wid = threadIdx.x >> 5, lane = threadIdx.x & 31;
  const int tile = blockIdx.x * 8 + wid;
  if (tile >= total_tiles) return;
  const int mg = tile / ntiles_n, nt = tile % ntiles_n;
  v8f acc[MBLK];
#pragma unroll
  for (int i = 0; i < MBLK; ++i) acc[i] = {};
  for (int k = 0; k < K; k += 32) {
    v16bf b = load_frag(Bt, K, nt * 16, k, lane);
    __builtin_prefetch(Bt + (size_t)(nt * 16 + (lane & 15)) * K + k + 128, 0, 2);
#pragma unroll
    for (int i = 0; i < MBLK; ++i) {
      v16bf a = load_frag(A, K, (mg * MBLK + i) * 16, k, lane);
      acc[i] = WMMA_BF16(a, b, acc[i]);
    }
  }
  const int n    = nt * 16 + (lane & 15);
  const int roff = (lane >> 4) << 3;
  const float bv = bias ? bias[n] : 0.f;
#pragma unroll
  for (int i = 0; i < MBLK; ++i) {
    const int mbase = (mg * MBLK + i) * 16 + roff;
#pragma unroll
    for (int r = 0; r < 8; ++r)
      out[(size_t)(mbase + r) * out_ld + n] = acc[i][r] + bv;
  }
}

// ---------------------------------------------------------------------------
// Fused LSTM gate GEMM:  g = X[256,K] @ Wt[4096,K]^T + b ; gates -> h,c.
// Each wave: 2 m-tiles x (i,f,o,g) n-tiles = 8 accumulators; per k-step
// 2 A-frags + 4 B-frags feed 8 WMMAs (4x A-reuse, 2x B-reuse).  Epilogue
// applies sigmoid/tanh, updates c in place, and emits fp32 + bf16 copies
// of h/c for the downstream GEMMs.
// ---------------------------------------------------------------------------
__global__ __launch_bounds__(256)
void lstm_gate_kernel(const __bf16* __restrict__ X, int K,
                      const __bf16* __restrict__ Wt,
                      const float* __restrict__ bias,
                      float* __restrict__ c_state,      // in/out [256,1024]
                      float* __restrict__ h_f32,        // out    [256,1024]
                      __bf16* __restrict__ h_bf16, long h_ld,   // optional
                      __bf16* __restrict__ c_bf16) {            // optional
  const int wid = threadIdx.x >> 5, lane = threadIdx.x & 31;
  const int tile = blockIdx.x * 8 + wid;          // 512 wave-tiles total
  const int mg = tile >> 6, nt = tile & 63;       // mg: 0..7 (2 m-tiles each)
  v8f acc[2][4];
#pragma unroll
  for (int i = 0; i < 2; ++i)
#pragma unroll
    for (int g = 0; g < 4; ++g) acc[i][g] = {};
  for (int k = 0; k < K; k += 32) {
    v16bf a0 = load_frag(X, K, (mg * 2 + 0) * 16, k, lane);
    v16bf a1 = load_frag(X, K, (mg * 2 + 1) * 16, k, lane);
    __builtin_prefetch(Wt + (size_t)(nt * 16 + (lane & 15)) * K + k + 128, 0, 2);
#pragma unroll
    for (int g = 0; g < 4; ++g) {
      v16bf b = load_frag(Wt, K, g * HH + nt * 16, k, lane);
      acc[0][g] = WMMA_BF16(a0, b, acc[0][g]);
      acc[1][g] = WMMA_BF16(a1, b, acc[1][g]);
    }
  }
  const int n    = nt * 16 + (lane & 15);
  const int roff = (lane >> 4) << 3;
  const float bvi = bias[n], bvf = bias[HH + n], bvo = bias[2 * HH + n], bvg = bias[3 * HH + n];
#pragma unroll
  for (int i = 0; i < 2; ++i) {
    const int mbase = (mg * 2 + i) * 16 + roff;
#pragma unroll
    for (int r = 0; r < 8; ++r) {
      const int m = mbase + r;
      const size_t o = (size_t)m * HH + n;
      const float gi = acc[i][0][r] + bvi, gf = acc[i][1][r] + bvf;
      const float go = acc[i][2][r] + bvo, gg = acc[i][3][r] + bvg;
      const float cn = sigm(gf) * c_state[o] + sigm(gi) * tanhf(gg);
      const float hn = sigm(go) * tanhf(cn);
      c_state[o] = cn;
      h_f32[o]   = hn;
      if (h_bf16) h_bf16[(size_t)m * h_ld + n] = (__bf16)hn;
      if (c_bf16) c_bf16[o] = (__bf16)cn;
    }
  }
}

// ---------------------------------------------------------------------------
// BN + LeakyReLU + max over L, then bf16 into x2_0[:, 0:512]
// ---------------------------------------------------------------------------
__global__ void pool_kernel(const float* __restrict__ feat,
                            const float* __restrict__ cb,
                            const float* __restrict__ gamma, const float* __restrict__ beta,
                            const float* __restrict__ mean,  const float* __restrict__ var,
                            __bf16* __restrict__ x2_0) {
  int idx = blockIdx.x * blockDim.x + threadIdx.x;
  if (idx >= BB * MM) return;
  int b = idx / MM, m = idx % MM;
  float inv = gamma[m] * rsqrtf(var[m] + 1e-5f);
  float shift = (cb[m] - mean[m]) * inv + beta[m];
  float mx = -3.4e38f;
#pragma unroll
  for (int l = 0; l < LL; ++l) {
    float f = feat[(size_t)(b * LL + l) * MM + m] * inv + shift;
    f = (f > 0.f) ? f : 0.01f * f;
    mx = fmaxf(mx, f);
  }
  x2_0[(size_t)b * K0 + m] = (__bf16)mx;
}

// Per-step: token embedding + previous h0 (fp32) -> bf16 sections of x2_0
__global__ void prestep_kernel(const float* __restrict__ emb, const int* __restrict__ xTok,
                               const float* __restrict__ h0f, __bf16* __restrict__ x2_0, int t) {
  int idx = blockIdx.x * blockDim.x + threadIdx.x;
  if (idx >= BB * (EE + HH)) return;
  int b = idx / (EE + HH), j = idx % (EE + HH);
  if (j < EE) {
    int tok = xTok[b * TT + t];
    x2_0[(size_t)b * K0 + MM + j] = (__bf16)emb[(size_t)tok * EE + j];
  } else {
    int h = j - EE;
    x2_0[(size_t)b * K0 + MM + EE + h] = (__bf16)h0f[(size_t)b * HH + h];
  }
}

// Attention head + copy prev h1 into x2_1.  One block per batch row.
__global__ void att_kernel(const float* __restrict__ h0f, const float* __restrict__ c0f,
                           const float* __restrict__ w1, const float* __restrict__ b1,
                           const float* __restrict__ w2, const float* __restrict__ b2,
                           const float* __restrict__ h1f, __bf16* __restrict__ x2_1) {
  __shared__ float sh[50];
  __shared__ float so[10];
  const int b = blockIdx.x, tid = threadIdx.x;
  if (tid < 50) {
    float acc = b1[tid];
    const float* h0p = h0f + (size_t)b * HH;
    const float* c0p = c0f + (size_t)b * HH;
    for (int k = 0; k < HH; ++k) acc += h0p[k] * w1[(size_t)k * 50 + tid];
    for (int k = 0; k < HH; ++k) acc += c0p[k] * w1[(size_t)(HH + k) * 50 + tid];
    sh[tid] = (acc > 0.f) ? acc : 0.01f * acc;
  }
  __syncthreads();
  if (tid < 10) {
    float acc = b2[tid];
    for (int j = 0; j < 50; ++j) acc += sh[j] * w2[j * 10 + tid];
    so[tid] = acc;
  }
  __syncthreads();
  if (tid < 10) {
    float mx = so[0];
    for (int j = 1; j < 10; ++j) mx = fmaxf(mx, so[j]);
    float s = 0.f;
    for (int j = 0; j < 10; ++j) s += __expf(so[j] - mx);
    x2_1[(size_t)b * K1P + HH + tid] = (__bf16)(__expf(so[tid] - mx) / s);
  }
  for (int j = tid; j < HH; j += blockDim.x)
    x2_1[(size_t)b * K1P + HH + 10 + j] = (__bf16)h1f[(size_t)b * HH + j];
}

// Final state tensor: [2, B, 2H] = stack(cat(h0,c0), cat(h1,c1))
__global__ void state_out_kernel(const float* __restrict__ h0, const float* __restrict__ c0,
                                 const float* __restrict__ h1, const float* __restrict__ c1,
                                 float* __restrict__ out) {
  int idx = blockIdx.x * blockDim.x + threadIdx.x;
  if (idx >= 2 * BB * 2 * HH) return;
  int layer = idx / (BB * 2 * HH);
  int rem   = idx % (BB * 2 * HH);
  int b = rem / (2 * HH), j = rem % (2 * HH);
  const float* src = layer ? (j < HH ? h1 : c1) : (j < HH ? h0 : c0);
  out[idx] = src[(size_t)b * HH + (j & (HH - 1))];
}

// ---------------------------------------------------------------------------
static inline int grid1(size_t n) { return (int)((n + 255) / 256); }

extern "C" void kernel_launch(void* const* d_in, const int* in_sizes, int n_in,
                              void* d_out, int out_size, void* d_ws, size_t ws_size,
                              hipStream_t stream) {
  const float* cnn    = (const float*)d_in[0];
  const int*   xTok   = (const int*)  d_in[1];
  const float* emb    = (const float*)d_in[3];
  const float* conv_w = (const float*)d_in[4];
  const float* conv_b = (const float*)d_in[5];
  const float* gamma  = (const float*)d_in[6];
  const float* beta   = (const float*)d_in[7];
  const float* mean   = (const float*)d_in[8];
  const float* var    = (const float*)d_in[9];
  const float* aw1    = (const float*)d_in[10];
  const float* ab1    = (const float*)d_in[11];
  const float* aw2    = (const float*)d_in[12];
  const float* ab2    = (const float*)d_in[13];
  const float* W0     = (const float*)d_in[14];
  const float* b0     = (const float*)d_in[15];
  const float* W1     = (const float*)d_in[16];
  const float* b1     = (const float*)d_in[17];
  const float* outw   = (const float*)d_in[18];
  const float* outb   = (const float*)d_in[19];
  float* out = (float*)d_out;

  // workspace carve-up (256B aligned regions)
  char* ws = (char*)d_ws;
  size_t off = 0;
  auto alloc = [&](size_t bytes) { size_t o = off; off = (off + bytes + 255) & ~(size_t)255; return o; };
  __bf16* W0t   = (__bf16*)(ws + alloc((size_t)4 * HH * K0  * 2));   // [4096,2048]
  __bf16* W1t   = (__bf16*)(ws + alloc((size_t)4 * HH * K1P * 2));   // [4096,2080]
  __bf16* OWt   = (__bf16*)(ws + alloc((size_t)VV * HH * 2));        // [10000,1024]
  __bf16* CWt   = (__bf16*)(ws + alloc((size_t)MM * NFF * 2));       // [512,2048]
  __bf16* Aconv = (__bf16*)(ws + alloc((size_t)BB * LL * NFF * 2));  // [2560,2048]
  float*  feat  = (float*) (ws + alloc((size_t)BB * LL * MM * 4));   // [2560,512]
  __bf16* x2_0  = (__bf16*)(ws + alloc((size_t)BB * K0  * 2));       // [256,2048]
  __bf16* x2_1  = (__bf16*)(ws + alloc((size_t)BB * K1P * 2));       // [256,2080]
  float*  h0f   = (float*) (ws + alloc((size_t)BB * HH * 4));
  float*  c0f   = (float*) (ws + alloc((size_t)BB * HH * 4));
  float*  h1f   = (float*) (ws + alloc((size_t)BB * HH * 4));
  float*  c1f   = (float*) (ws + alloc((size_t)BB * HH * 4));
  __bf16* c1bf  = (__bf16*)(ws + alloc((size_t)BB * HH * 2));

  // zero recurrent state + activation staging (graph-capturable memset nodes)
  hipMemsetAsync(x2_0, 0, (size_t)BB * K0  * 2, stream);
  hipMemsetAsync(x2_1, 0, (size_t)BB * K1P * 2, stream);
  hipMemsetAsync(h0f,  0, (size_t)BB * HH * 4, stream);
  hipMemsetAsync(c0f,  0, (size_t)BB * HH * 4, stream);
  hipMemsetAsync(h1f,  0, (size_t)BB * HH * 4, stream);
  hipMemsetAsync(c1f,  0, (size_t)BB * HH * 4, stream);

  // one-time bf16 weight prep (L2-resident thereafter: ~70 MB << 192 MB)
  transpose_to_bf16_kernel<<<grid1((size_t)4 * HH * K0),  256, 0, stream>>>(W0,   W0t, K0,   4 * HH, K0);
  transpose_to_bf16_kernel<<<grid1((size_t)4 * HH * K1P), 256, 0, stream>>>(W1,   W1t, 2058, 4 * HH, K1P);
  transpose_to_bf16_kernel<<<grid1((size_t)VV * HH),      256, 0, stream>>>(outw, OWt, HH,   VV,     HH);
  f32_to_bf16_kernel<<<grid1((size_t)MM * NFF), 256, 0, stream>>>(conv_w, CWt, (size_t)MM * NFF);
  conv_a_prep_kernel<<<grid1((size_t)BB * LL * NFF), 256, 0, stream>>>(cnn, Aconv);

  // conv GEMM (2560 x 2048 x 512), MBLK=4: 40 m-groups x 32 n-tiles
  gemm_bias_kernel<4><<<(40 * 32) / 8, 256, 0, stream>>>(Aconv, CWt, nullptr, feat,
                                                         32, 40 * 32, NFF, MM);
  pool_kernel<<<grid1((size_t)BB * MM), 256, 0, stream>>>(feat, conv_b, gamma, beta, mean, var, x2_0);

  for (int t = 0; t < TT; ++t) {
    // token embedding + h0_{t-1} -> x2_0 (bf16)
    prestep_kernel<<<grid1((size_t)BB * (EE + HH)), 256, 0, stream>>>(emb, xTok, h0f, x2_0, t);
    // layer 0: 256 x 2048 x 4096 fused-gate GEMM; h0 bf16 -> x2_1[:,0:1024]
    lstm_gate_kernel<<<64, 256, 0, stream>>>(x2_0, K0, W0t, b0, c0f, h0f, x2_1, K1P, nullptr);
    // attention (writes x2_1[:,1024:1034]) + h1_{t-1} -> x2_1[:,1034:2058]
    att_kernel<<<BB, 128, 0, stream>>>(h0f, c0f, aw1, ab1, aw2, ab2, h1f, x2_1);
    // layer 1: 256 x 2080 x 4096 fused-gate GEMM; c1 bf16 for logits
    lstm_gate_kernel<<<64, 256, 0, stream>>>(x2_1, K1P, W1t, b1, c1f, h1f, nullptr, 0, c1bf);
    // logits: 256 x 1024 x 10000, MBLK=4: 4 m-groups x 625 n-tiles -> d_out[:, t, :]
    gemm_bias_kernel<4><<<(4 * 625 + 7) / 8, 256, 0, stream>>>(c1bf, OWt, outb,
        out + (size_t)t * VV, 625, 4 * 625, HH, (long)TT * VV);
  }

  // final state [2, B, 2H]
  state_out_kernel<<<grid1((size_t)2 * BB * 2 * HH), 256, 0, stream>>>(
      h0f, c0f, h1f, c1f, out + (size_t)BB * TT * VV);
}
